// Transformer_19911468384330
// MI455X (gfx1250) — compile-verified, matrix operand order checked by reading, not run on hardware
//
#include <hip/hip_runtime.h>
#include <hip/hip_bf16.h>

// ---------------- model constants (from reference) ----------------
#define S_LEN   2048
#define D_MODEL 2048
#define NH      16
#define NKV     4
#define HDIM    128
#define KVD     (NKV * HDIM)   // 512
#define FF_DIM  5632
#define VOCAB   32000
#define N_LAYER 2
#define KSTEP   64             // GEMM K-step (two 32-deep WMMA sub-steps per stage)

typedef __attribute__((ext_vector_type(16))) __bf16          v16bf;
typedef __attribute__((ext_vector_type(8)))  float           v8f;
typedef __attribute__((ext_vector_type(8)))  unsigned int    v8u;
typedef __attribute__((ext_vector_type(8)))  unsigned short  us8;   // 16 bytes
typedef unsigned short u16;

// fp32 -> bf16 bits, round-to-nearest-even (only in bandwidth-bound converters)
__device__ __forceinline__ u16 bf_bits(float f) {
    unsigned u = __builtin_bit_cast(unsigned, f);
    return (u16)((u + 0x7FFFu + ((u >> 16) & 1u)) >> 16);
}
__device__ __forceinline__ float bf_f(u16 s) {
    unsigned u = ((unsigned)s) << 16;
    return __builtin_bit_cast(float, u);
}
__device__ __forceinline__ __bf16 us_bf(u16 s) { return __builtin_bit_cast(__bf16, s); }

// K index of element e (0..15) in a 16x32 bf16 WMMA A/B fragment (ISA 7.12.2):
// lanes 0-15 hold K {0..7,16..23}; lanes 16-31 hold K {8..15,24..31}.
__device__ __forceinline__ int kmap(int e, int kh) {
    return ((e < 8) ? 0 : 16) + kh * 8 + (e & 7);
}
// starting K of dword j (0..7) = elements {2j,2j+1} (always contiguous in K)
__device__ __forceinline__ int kpair(int j, int kh) {
    return ((j < 4) ? 0 : 16) + kh * 8 + ((j & 3) << 1);
}

// ---------------- generic bf16 WMMA GEMM ----------------
// C = A[M,K](bf16) * B(bf16) (+ Cin fp32 if ACC).  BT: B is [N,K] (lm_head).
// OUTBF: write bf16 C (feeds the next matmul), else fp32.
// 256 threads = 8 wave32, macro tile 64(M) x 128(N), K step 64, double-buffered
// async-to-LDS staging (ASYNCcnt) for the contiguous-copy paths.
template<bool BT, bool ACC, bool OUTBF>
__global__ __launch_bounds__(256) void gemm_bf16(
    void* __restrict__ Cv, const u16* __restrict__ A, const u16* __restrict__ B,
    const float* __restrict__ Cin, int M, int N, int K)
{
    // single LDS object so async-to-LDS byte offsets start at 0
    // stage = As 64x64 + Bs 128x64 (u16), double buffered
    constexpr int A_ELEMS  = 64 * KSTEP;
    constexpr int B_ELEMS  = 128 * KSTEP;
    constexpr int STAGE_EL = A_ELEMS + B_ELEMS;
    __shared__ u16 smem[2 * STAGE_EL];

    const int t = threadIdx.x;
    const int lane = t & 31, wid = t >> 5;
    const int waveM = wid >> 2, waveN = wid & 3;          // 2 x 4 wave grid
    const int mBase = blockIdx.y * 64, nBase = blockIdx.x * 128;
    const int m = lane & 15, kh = lane >> 4;

    v8f acc[2][2] = {};

    // ---- staging helper (issues async copies / register copies for one K-step) ----
    auto stage = [&](int buf, int kt) {
        const unsigned ldsBase = (unsigned)(buf * STAGE_EL * 2); // bytes
        { // A tile 64x64: two async b128 per thread (contiguous both sides)
            int r = t >> 2, c0 = (t & 3) * 16;
            unsigned lo = ldsBase + (unsigned)((r * KSTEP + c0) * 2);
            unsigned long long ga =
                (unsigned long long)(size_t)(A + (size_t)(mBase + r) * K + kt + c0);
            asm volatile("global_load_async_to_lds_b128 %0, %1, off"
                         :: "v"(lo), "v"(ga) : "memory");
            asm volatile("global_load_async_to_lds_b128 %0, %1, off offset:16"
                         :: "v"(lo), "v"(ga) : "memory");
        }
        if (BT) { // B is [N,K]: rows map onto Bs columns -> four async b128 per thread
            int n = t >> 1, k0 = (t & 1) * 32;
            unsigned lo = ldsBase + (unsigned)((A_ELEMS + n * KSTEP + k0) * 2);
            unsigned long long ga =
                (unsigned long long)(size_t)(B + (size_t)(nBase + n) * K + kt + k0);
            asm volatile("global_load_async_to_lds_b128 %0, %1, off"
                         :: "v"(lo), "v"(ga) : "memory");
            asm volatile("global_load_async_to_lds_b128 %0, %1, off offset:16"
                         :: "v"(lo), "v"(ga) : "memory");
            asm volatile("global_load_async_to_lds_b128 %0, %1, off offset:32"
                         :: "v"(lo), "v"(ga) : "memory");
            asm volatile("global_load_async_to_lds_b128 %0, %1, off offset:48"
                         :: "v"(lo), "v"(ga) : "memory");
        } else {  // B is [K,N]: contiguous global reads, scattered LDS column stores
            u16* Bs = smem + buf * STAGE_EL + A_ELEMS;
            int k = t >> 2, c0 = (t & 3) * 32;
            const u16* src = B + (size_t)(kt + k) * N + nBase + c0;
            us8 w0 = *(const us8*)(src);
            us8 w1 = *(const us8*)(src + 8);
            us8 w2 = *(const us8*)(src + 16);
            us8 w3 = *(const us8*)(src + 24);
            #pragma unroll
            for (int j = 0; j < 8; ++j) {
                Bs[(c0 + j)      * KSTEP + k] = w0[j];
                Bs[(c0 + 8 + j)  * KSTEP + k] = w1[j];
                Bs[(c0 + 16 + j) * KSTEP + k] = w2[j];
                Bs[(c0 + 24 + j) * KSTEP + k] = w3[j];
            }
            if (kt + KSTEP < K) __builtin_prefetch(src + (size_t)KSTEP * N, 0, 1);
        }
    };

    stage(0, 0); // prologue: fill buffer 0

    for (int kt = 0; kt < K; kt += KSTEP) {
        const int buf = (kt / KSTEP) & 1;
        asm volatile("s_wait_asynccnt 0" ::: "memory"); // current stage landed in LDS
        __syncthreads();
        if (kt + KSTEP < K) stage(buf ^ 1, kt + KSTEP); // overlap next DMA with compute

        const u16* As = smem + buf * STAGE_EL;
        const u16* Bs = As + A_ELEMS;
        #pragma unroll
        for (int ks = 0; ks < KSTEP; ks += 32) {
            v16bf af[2], bfr[2];
            #pragma unroll
            for (int mi = 0; mi < 2; ++mi) {
                int r0 = waveM * 32 + mi * 16 + m;
                v8u u;
                #pragma unroll
                for (int j = 0; j < 8; ++j)
                    u[j] = *(const unsigned*)(&As[r0 * KSTEP + ks + kpair(j, kh)]);
                af[mi] = __builtin_bit_cast(v16bf, u);
            }
            #pragma unroll
            for (int ni = 0; ni < 2; ++ni) {
                int c0 = waveN * 32 + ni * 16 + m;
                v8u u;
                #pragma unroll
                for (int j = 0; j < 8; ++j)
                    u[j] = *(const unsigned*)(&Bs[c0 * KSTEP + ks + kpair(j, kh)]);
                bfr[ni] = __builtin_bit_cast(v16bf, u);
            }
            #pragma unroll
            for (int mi = 0; mi < 2; ++mi)
                #pragma unroll
                for (int ni = 0; ni < 2; ++ni)
                    acc[mi][ni] = __builtin_amdgcn_wmma_f32_16x16x32_bf16(
                        false, af[mi], false, bfr[ni], (short)0, acc[mi][ni], false, false);
        }
        __syncthreads();
    }

    // epilogue: C layout -> VGPR r holds M = r + 8*(lane>=16), N = lane&15
    #pragma unroll
    for (int mi = 0; mi < 2; ++mi)
        #pragma unroll
        for (int ni = 0; ni < 2; ++ni)
            #pragma unroll
            for (int r = 0; r < 8; ++r) {
                int row = mBase + waveM * 32 + mi * 16 + kh * 8 + r;
                int col = nBase + waveN * 32 + ni * 16 + m;
                size_t o = (size_t)row * N + col;
                float v = acc[mi][ni][r];
                if (ACC) v += Cin[o];
                if (OUTBF) ((u16*)Cv)[o]   = bf_bits(v);
                else       ((float*)Cv)[o] = v;
            }
}

// ---------------- attention (WMMA QK^T + softmax + WMMA PV) ----------------
// grid = (S/16, H); block = 256 (8 waves). q/k/v pre-converted bf16.
// Faithful to reference: tril mask MULTIPLIES scores (masked -> 0.0), softmax over full S.
// LDS: 16*S fp32 scores + 16*S bf16 probs (+ reductions) ~ 198 KB (within 320 KB/WGP).
__global__ __launch_bounds__(256) void attn_kernel(
    u16* __restrict__ y, const u16* __restrict__ q,
    const u16* __restrict__ kbuf, const u16* __restrict__ vbuf)
{
    extern __shared__ char smem_raw[];
    float* sc     = (float*)smem_raw;        // 16 * S_LEN fp32 scores
    float* red    = sc + 16 * S_LEN;         // 256
    float* rowmax = red + 256;               // 16
    float* rowsum = rowmax + 16;             // 16
    u16*   pbf    = (u16*)(rowsum + 16);     // 16 * S_LEN bf16 probs

    const int t = threadIdx.x, lane = t & 31, wid = t >> 5;
    const int q0 = blockIdx.x * 16, h = blockIdx.y;
    const int kvh = h / (NH / NKV);
    const int m = lane & 15, kh = lane >> 4;
    const float scale = 0.08838834764831844f; // 1/sqrt(128)

    // ---- Phase 1: scores 16 x S via WMMA; dword fragment loads from bf16 global ----
    v16bf aq[4];
    #pragma unroll
    for (int ks = 0; ks < 4; ++ks) {
        v8u u;
        #pragma unroll
        for (int j = 0; j < 8; ++j)
            u[j] = *(const unsigned*)(q + (size_t)(q0 + m) * D_MODEL + h * HDIM + ks * 32 + kpair(j, kh));
        aq[ks] = __builtin_bit_cast(v16bf, u);
    }

    for (int kt = wid; kt < S_LEN / 16; kt += 8) {
        v8f s = {};
        #pragma unroll
        for (int ks = 0; ks < 4; ++ks) {
            v8u u; // B = K^T: col n = lane&15 -> key kt*16+n; K pairs contiguous in kbuf row
            #pragma unroll
            for (int j = 0; j < 8; ++j)
                u[j] = *(const unsigned*)(kbuf + (size_t)(kt * 16 + m) * KVD + kvh * HDIM + ks * 32 + kpair(j, kh));
            v16bf bk = __builtin_bit_cast(v16bf, u);
            s = __builtin_amdgcn_wmma_f32_16x16x32_bf16(false, aq[ks], false, bk, (short)0, s, false, false);
        }
        #pragma unroll
        for (int r = 0; r < 8; ++r) {
            int mr  = kh * 8 + r;
            int key = kt * 16 + m;
            float val = s[r] * scale;
            if (key > q0 + mr) val = 0.0f;   // reference: scores * tril mask
            sc[mr * S_LEN + key] = val;
        }
    }
    __syncthreads();

    // ---- Phase 2: softmax per row over full S; write probs once as bf16 ----
    {
        int row = t >> 4, sub = t & 15;
        float mx = -3.0e38f;
        for (int j = sub; j < S_LEN; j += 16) mx = fmaxf(mx, sc[row * S_LEN + j]);
        red[row * 16 + sub] = mx;
        __syncthreads();
        if (sub == 0) {
            float v = red[row * 16];
            for (int j = 1; j < 16; ++j) v = fmaxf(v, red[row * 16 + j]);
            rowmax[row] = v;
        }
        __syncthreads();
        float sm = 0.0f, rm = rowmax[row];
        for (int j = sub; j < S_LEN; j += 16) {
            float e = __expf(sc[row * S_LEN + j] - rm);
            pbf[row * S_LEN + j] = bf_bits(e);
            sm += e;
        }
        red[row * 16 + sub] = sm;
        __syncthreads();
        if (sub == 0) {
            float v = 0.0f;
            for (int j = 0; j < 16; ++j) v += red[row * 16 + j];
            rowsum[row] = v;
        }
        __syncthreads();
    }

    // ---- Phase 3: O = P * V; wave wid owns 16 head dims ----
    {
        int c0 = wid * 16;
        v8f o = {};
        for (int ks = 0; ks < S_LEN / 32; ++ks) {
            v8u ua;
            #pragma unroll
            for (int j = 0; j < 8; ++j)
                ua[j] = *(const unsigned*)(pbf + m * S_LEN + ks * 32 + kpair(j, kh));
            v16bf ap = __builtin_bit_cast(v16bf, ua);
            v16bf bv; // V rows = keys (stride KVD) -> per-element, lane-coalesced loads
            #pragma unroll
            for (int e = 0; e < 16; ++e)
                bv[e] = us_bf(vbuf[(size_t)(ks * 32 + kmap(e, kh)) * KVD + kvh * HDIM + c0 + m]);
            o = __builtin_amdgcn_wmma_f32_16x16x32_bf16(false, ap, false, bv, (short)0, o, false, false);
        }
        #pragma unroll
        for (int r = 0; r < 8; ++r) {
            int mr = kh * 8 + r;
            y[(size_t)(q0 + mr) * D_MODEL + h * HDIM + c0 + m] = bf_bits(o[r] / rowsum[mr]);
        }
    }
}

// ---------------- elementwise / small kernels ----------------
__global__ __launch_bounds__(256) void cvt_bf16_kernel(
    u16* __restrict__ dst, const float* __restrict__ src, size_t n)
{
    size_t i = ((size_t)blockIdx.x * 256 + threadIdx.x) * 4;
    if (i + 3 < n) {
        float4 v = *(const float4*)(src + i);
        u16 o0 = bf_bits(v.x), o1 = bf_bits(v.y), o2 = bf_bits(v.z), o3 = bf_bits(v.w);
        *(unsigned*)(dst + i)     = (unsigned)o0 | ((unsigned)o1 << 16);
        *(unsigned*)(dst + i + 2) = (unsigned)o2 | ((unsigned)o3 << 16);
    }
}

__global__ __launch_bounds__(256) void embed_kernel(
    float* __restrict__ x, const float* __restrict__ emb, const int* __restrict__ idx)
{
    int s = blockIdx.x;
    int tok = idx[s];
    for (int d = threadIdx.x; d < D_MODEL; d += 256)
        x[(size_t)s * D_MODEL + d] = emb[(size_t)tok * D_MODEL + d];
}

// RMSNorm: fp32 in -> bf16 out (feeds GEMM A operand)
__global__ __launch_bounds__(256) void rmsnorm_kernel(
    u16* __restrict__ out, const float* __restrict__ x, const float* __restrict__ g)
{
    __shared__ float red[256];
    int row = blockIdx.x;
    const float* xr = x + (size_t)row * D_MODEL;
    float s = 0.0f;
    for (int d = threadIdx.x; d < D_MODEL; d += 256) { float v = xr[d]; s += v * v; }
    red[threadIdx.x] = s;
    __syncthreads();
    for (int st = 128; st > 0; st >>= 1) {
        if (threadIdx.x < st) red[threadIdx.x] += red[threadIdx.x + st];
        __syncthreads();
    }
    float inv = rsqrtf(red[0] / (float)D_MODEL + 1e-5f);
    for (int d = threadIdx.x; d < D_MODEL; d += 256)
        out[(size_t)row * D_MODEL + d] = bf_bits(xr[d] * inv * g[d]);
}

// RoPE in-place on bf16 q/k
__global__ void rope_kernel(u16* __restrict__ x, int nh)
{
    int i = blockIdx.x * blockDim.x + threadIdx.x;
    int total = S_LEN * nh * (HDIM / 2);
    if (i >= total) return;
    int d  = i % (HDIM / 2);
    int hh = (i / (HDIM / 2)) % nh;
    int s  = i / ((HDIM / 2) * nh);
    float inv = __expf(-(2.0f * (float)d / (float)HDIM) * __logf(10000.0f));
    float f = (float)s * inv;
    float c = cosf(f), sn = sinf(f);
    size_t base = (size_t)s * nh * HDIM + (size_t)hh * HDIM;
    float x1 = bf_f(x[base + d]), x2 = bf_f(x[base + d + HDIM / 2]);
    x[base + d]            = bf_bits(x1 * c - x2 * sn);
    x[base + d + HDIM / 2] = bf_bits(x2 * c + x1 * sn);
}

// g = silu(u) * g  (bf16 in/out)
__global__ void silu_mul_kernel(u16* __restrict__ g, const u16* __restrict__ u, int n)
{
    int i = blockIdx.x * blockDim.x + threadIdx.x;
    if (i < n) {
        float uv = bf_f(u[i]);
        float s = uv / (1.0f + __expf(-uv));
        g[i] = bf_bits(s * bf_f(g[i]));
    }
}

// ---------------- host orchestration ----------------
extern "C" void kernel_launch(void* const* d_in, const int* in_sizes, int n_in,
                              void* d_out, int out_size, void* d_ws, size_t ws_size,
                              hipStream_t stream)
{
    (void)in_sizes; (void)n_in; (void)out_size; (void)ws_size;
    const int*   idx  = (const int*)  d_in[0];
    const float* emb  = (const float*)d_in[1];
    const float* wq   = (const float*)d_in[2];
    const float* wk   = (const float*)d_in[3];
    const float* wv   = (const float*)d_in[4];
    const float* wo   = (const float*)d_in[5];
    const float* ln1  = (const float*)d_in[6];
    const float* ln2  = (const float*)d_in[7];
    const float* wg   = (const float*)d_in[8];
    const float* wu   = (const float*)d_in[9];
    const float* wd   = (const float*)d_in[10];
    const float* fing = (const float*)d_in[11];
    const float* lmh  = (const float*)d_in[12];
    float* out = (float*)d_out;

    // workspace layout: fp32 residual stream first, then bf16 (u16) buffers
    const size_t SD = (size_t)S_LEN * D_MODEL;
    const size_t SKV = (size_t)S_LEN * KVD;
    const size_t SFF = (size_t)S_LEN * FF_DIM;
    const size_t WMAX = (size_t)VOCAB * D_MODEL; // >= D*FF

    float* x    = (float*)d_ws;                 // SD fp32
    u16*  base16 = (u16*)(x + SD);
    u16*  hbf   = base16;                       // SD
    u16*  qbf   = hbf  + SD;                    // SD
    u16*  kbf   = qbf  + SD;                    // SKV
    u16*  vbf   = kbf  + SKV;                   // SKV
    u16*  tmpbf = vbf  + SKV;                   // SD (attn out)
    u16*  gbf   = tmpbf + SD;                   // SFF
    u16*  ubf   = gbf  + SFF;                   // SFF
    u16*  wbf   = ubf  + SFF;                   // WMAX (per-use weight staging)

    const dim3 blk(256);
    const dim3 gD (D_MODEL / 128, S_LEN / 64);
    const dim3 gKV(KVD    / 128, S_LEN / 64);
    const dim3 gFF(FF_DIM / 128, S_LEN / 64);
    const dim3 gV (VOCAB  / 128, S_LEN / 64);
    const size_t attnSmem = (size_t)(16 * S_LEN) * 4 + 288 * 4 + (size_t)(16 * S_LEN) * 2;

    auto cvt = [&](u16* dst, const float* src, size_t n) {
        cvt_bf16_kernel<<<(unsigned)((n / 4 + 255) / 256), blk, 0, stream>>>(dst, src, n);
    };

    embed_kernel<<<S_LEN, blk, 0, stream>>>(x, emb, idx);

    for (int l = 0; l < N_LAYER; ++l) {
        const float* lwq = wq + (size_t)l * D_MODEL * D_MODEL;
        const float* lwk = wk + (size_t)l * D_MODEL * KVD;
        const float* lwv = wv + (size_t)l * D_MODEL * KVD;
        const float* lwo = wo + (size_t)l * D_MODEL * D_MODEL;
        const float* lg1 = ln1 + (size_t)l * D_MODEL;
        const float* lg2 = ln2 + (size_t)l * D_MODEL;
        const float* lwg = wg + (size_t)l * D_MODEL * FF_DIM;
        const float* lwu = wu + (size_t)l * D_MODEL * FF_DIM;
        const float* lwd = wd + (size_t)l * FF_DIM * D_MODEL;

        rmsnorm_kernel<<<S_LEN, blk, 0, stream>>>(hbf, x, lg1);

        cvt(wbf, lwq, (size_t)D_MODEL * D_MODEL);
        gemm_bf16<false,false,true><<<gD,  blk, 0, stream>>>(qbf, hbf, wbf, nullptr, S_LEN, D_MODEL, D_MODEL);
        cvt(wbf, lwk, (size_t)D_MODEL * KVD);
        gemm_bf16<false,false,true><<<gKV, blk, 0, stream>>>(kbf, hbf, wbf, nullptr, S_LEN, KVD, D_MODEL);
        cvt(wbf, lwv, (size_t)D_MODEL * KVD);
        gemm_bf16<false,false,true><<<gKV, blk, 0, stream>>>(vbf, hbf, wbf, nullptr, S_LEN, KVD, D_MODEL);

        {
            int nq = S_LEN * NH  * (HDIM / 2);
            int nk = S_LEN * NKV * (HDIM / 2);
            rope_kernel<<<(nq + 255) / 256, blk, 0, stream>>>(qbf, NH);
            rope_kernel<<<(nk + 255) / 256, blk, 0, stream>>>(kbf, NKV);
        }

        attn_kernel<<<dim3(S_LEN / 16, NH), blk, attnSmem, stream>>>(tmpbf, qbf, kbf, vbf);

        // x += attn @ wo (fused residual, fp32 out)
        cvt(wbf, lwo, (size_t)D_MODEL * D_MODEL);
        gemm_bf16<false,true,false><<<gD, blk, 0, stream>>>(x, tmpbf, wbf, x, S_LEN, D_MODEL, D_MODEL);

        rmsnorm_kernel<<<S_LEN, blk, 0, stream>>>(hbf, x, lg2);
        cvt(wbf, lwg, (size_t)D_MODEL * FF_DIM);
        gemm_bf16<false,false,true><<<gFF, blk, 0, stream>>>(gbf, hbf, wbf, nullptr, S_LEN, FF_DIM, D_MODEL);
        cvt(wbf, lwu, (size_t)D_MODEL * FF_DIM);
        gemm_bf16<false,false,true><<<gFF, blk, 0, stream>>>(ubf, hbf, wbf, nullptr, S_LEN, FF_DIM, D_MODEL);
        silu_mul_kernel<<<((S_LEN * FF_DIM) + 255) / 256, blk, 0, stream>>>(gbf, ubf, S_LEN * FF_DIM);
        // x += (silu(up)*gate) @ w_down (fused residual)
        cvt(wbf, lwd, (size_t)FF_DIM * D_MODEL);
        gemm_bf16<false,true,false><<<gD, blk, 0, stream>>>(x, gbf, wbf, x, S_LEN, D_MODEL, FF_DIM);
    }

    rmsnorm_kernel<<<S_LEN, blk, 0, stream>>>(hbf, x, fing);
    // logits = h @ lm_head.T  (lm_head stored [V, D] -> BT path, fp32 out)
    cvt(wbf, lmh, WMAX);
    gemm_bf16<true,false,false><<<gV, blk, 0, stream>>>(out, hbf, wbf, nullptr, S_LEN, VOCAB, D_MODEL);
}
